// MambaSequenceEncoder_37666863186329
// MI455X (gfx1250) — matrix operand order repaired and checked by reading, not compiled, not run on hardware
//
#include <hip/hip_runtime.h>
#include <hip/hip_bf16.h>

// ---------------- problem constants ----------------
#define BB   16
#define LL   4096
#define DM   256
#define NLAY 4
#define DS   16
#define DC   4
#define DI   512            // DM * EX
#define MR   (BB * LL)      // 65536 rows
#define CH   64             // scan chunk length
#define NCH  (LL / CH)      // 64 chunks

typedef __bf16 bf16;
typedef __attribute__((ext_vector_type(16))) __bf16 bf16x16;
typedef __attribute__((ext_vector_type(8)))  __bf16 bf16x8;
typedef __attribute__((ext_vector_type(8)))  float  f32x8;

// ---------------- weight fragment packing ----------------
// Fragment fi = nt*KT + kt holds a 32(K)x16(N) bf16 tile as a WMMA B operand, contiguous k order (same as the A load):
// lane 0-15: col = lane,    K = kt*32 + 0..15
// lane 16-31: col = lane-16, K = kt*32 + 16..31
__global__ void k_pack_w(const float* __restrict__ W, bf16* __restrict__ dst,
                         int K, int N) {
  int gt   = blockIdx.x * 256 + threadIdx.x;
  int lane = gt & 31;
  int fi   = gt >> 5;
  int KT = K >> 5, NT = N >> 4;
  if (fi >= KT * NT) return;
  int kt = fi % KT, nt = fi / KT;
  int n     = nt * 16 + (lane & 15);
  int kbase = kt * 32 + ((lane >> 4) << 4);
  bf16* d = dst + ((size_t)fi << 9) + (lane << 4);
#pragma unroll
  for (int j = 0; j < 16; ++j)
    d[j] = (bf16)W[(size_t)(kbase + j) * N + n];
}

// ---------------- WMMA GEMM: out[M,N] = A[M,K](bf16) * Wfrag(bf16) ----------------
// MODE 0: store f32
// MODE 1: +bias, softplus, store f32
// MODE 2: out += acc (residual accumulate)
// NSUB and MODE are compile-time so accumulators stay in statically-allocated
// even-aligned VGPR blocks (no movrel-indexed register file traffic).
template <int NSUB, int MODE>
__global__ void k_gemm_wmma(const bf16* __restrict__ A, const bf16* __restrict__ W,
                            float* __restrict__ out, const float* __restrict__ bias,
                            int N, int K) {
  int lane = threadIdx.x & 31;
  int wave = threadIdx.x >> 5;
  int Mbase = (blockIdx.y * 8 + wave) * 16;
  int Nbase = blockIdx.x * (NSUB * 16);
  int KT = K >> 5;
  int r  = lane & 15;
  int hi = lane >> 4;          // 0 or 1
  int koff = hi ? 16 : 0;

  f32x8 acc[NSUB];
#pragma unroll
  for (int s = 0; s < NSUB; ++s)
#pragma unroll
    for (int j = 0; j < 8; ++j) acc[s][j] = 0.0f;

  const bf16* arow  = A + (size_t)(Mbase + r) * K + koff;
  const bf16* wbase = W + ((size_t)((Nbase >> 4) * KT) << 9) + (lane << 4);

  for (int kt = 0; kt < KT; ++kt) {
    bf16x8 alo = *(const bf16x8*)(arow + kt * 32);
    bf16x8 ahi = *(const bf16x8*)(arow + kt * 32 + 8);
    bf16x16 a;
#pragma unroll
    for (int j = 0; j < 8; ++j) { a[j] = alo[j]; a[8 + j] = ahi[j]; }
#pragma unroll
    for (int s = 0; s < NSUB; ++s) {
      bf16x16 bfrag = *(const bf16x16*)(wbase + ((size_t)(s * KT + kt) << 9));
      acc[s] = __builtin_amdgcn_wmma_f32_16x16x32_bf16(
          false, a, false, bfrag, (short)0, acc[s], false, false);
    }
  }

#pragma unroll
  for (int s = 0; s < NSUB; ++s) {
    int col = Nbase + s * 16 + r;
#pragma unroll
    for (int v = 0; v < 8; ++v) {
      int row = Mbase + v + (hi << 3);
      size_t o = (size_t)row * N + col;
      float val = acc[s][v];
      if (MODE == 1) {
        val += bias[col];
        val = (val > 20.f) ? val : log1pf(__expf(val));
        out[o] = val;
      } else if (MODE == 2) {
        out[o] += val;
      } else {
        out[o] = val;
      }
    }
  }
}

// ---------------- embed: h[b,l,c] = sum_j x[b,j,l]*in_w[j,c] + in_b[c] ----------------
__global__ void k_embed(const float* __restrict__ x, const float* __restrict__ in_w,
                        const float* __restrict__ in_b, float* __restrict__ h) {
  size_t idx = (size_t)blockIdx.x * 256 + threadIdx.x;   // MR*DM total
  int c = idx & 255;
  size_t row = idx >> 8;
  int b = (int)(row >> 12);
  int l = (int)(row & 4095);
  float acc = in_b[c];
#pragma unroll
  for (int j = 0; j < 4; ++j)
    acc += x[((size_t)b * 4 + j) * LL + l] * in_w[j * DM + c];
  h[idx] = acc;
}

// ---------------- LayerNorm over DM=256 (one wave per row) ----------------
__global__ void k_layernorm(const float* __restrict__ in, const float* __restrict__ g,
                            const float* __restrict__ bta, float* __restrict__ outF,
                            bf16* __restrict__ outB) {
  int wave = threadIdx.x >> 5, lane = threadIdx.x & 31;
  size_t row = (size_t)blockIdx.x * 8 + wave;
  const float* p = in + row * DM;
  float v[8], s = 0.f, ss = 0.f;
#pragma unroll
  for (int j = 0; j < 8; ++j) {
    v[j] = p[j * 32 + lane];
    s += v[j]; ss += v[j] * v[j];
  }
#pragma unroll
  for (int off = 16; off > 0; off >>= 1) {
    s  += __shfl_xor(s,  off, 32);
    ss += __shfl_xor(ss, off, 32);
  }
  float mean = s * (1.f / DM);
  float var  = ss * (1.f / DM) - mean * mean;
  float rinv = rsqrtf(var + 1e-5f);
#pragma unroll
  for (int j = 0; j < 8; ++j) {
    int c = j * 32 + lane;
    float o = (v[j] - mean) * rinv * g[c] + bta[c];
    if (outF) outF[row * DM + c] = o;
    if (outB) outB[row * DM + c] = (bf16)o;
  }
}

// ---------------- causal depthwise conv (k=4) + SiLU ----------------
__global__ void k_conv_silu(const float* __restrict__ xz, const float* __restrict__ cw,
                            const float* __restrict__ cb, float* __restrict__ xs,
                            bf16* __restrict__ xsb) {
  size_t idx = (size_t)blockIdx.x * 256 + threadIdx.x;   // MR*DI total
  int d = idx & (DI - 1);
  size_t row = idx >> 9;
  int b = (int)(row >> 12);
  int l = (int)(row & 4095);
  float acc = cb[d];
#pragma unroll
  for (int k = 0; k < DC; ++k) {
    int lk = l - (DC - 1) + k;
    if (lk >= 0)
      acc += cw[d * DC + k] * xz[((size_t)b * LL + lk) * (2 * DI) + d];
  }
  acc = acc / (1.f + __expf(-acc));   // SiLU
  xs[idx]  = acc;
  xsb[idx] = (bf16)acc;
}

// ---------------- selective scan, pass 1: per-chunk local scan ----------------
__global__ void k_scan1(const float* __restrict__ dt, const float* __restrict__ xs,
                        const float* __restrict__ bc, const float* __restrict__ A_log,
                        float* __restrict__ chA, float* __restrict__ chH) {
  int gid = blockIdx.x * 256 + threadIdx.x;   // BB*NCH*DI
  int d  = gid & (DI - 1);
  int t2 = gid >> 9;
  int ck = t2 & (NCH - 1);
  int b  = t2 >> 6;
  float Ac[16], hst[16], ap[16];
#pragma unroll
  for (int n = 0; n < 16; ++n) {
    Ac[n] = -__expf(A_log[d * DS + n]);
    hst[n] = 0.f; ap[n] = 1.f;
  }
  int t0 = ck * CH;
  for (int tt = 0; tt < CH; ++tt) {
    size_t row = (size_t)b * LL + t0 + tt;
    float dtv = dt[row * DI + d];
    float xv  = xs[row * DI + d];
    float dtx = dtv * xv;
    const float* Bp = bc + row * 32;
#pragma unroll
    for (int n = 0; n < 16; ++n) {
      float e = __expf(Ac[n] * dtv);
      ap[n]  *= e;
      hst[n]  = e * hst[n] + dtx * Bp[n];
    }
  }
  size_t base = (((size_t)b * NCH + ck) * DI + d) * 16;
#pragma unroll
  for (int n = 0; n < 16; ++n) { chA[base + n] = ap[n]; chH[base + n] = hst[n]; }
}

// ---------------- scan pass 2: combine chunks; chA becomes chunk-start h0 ----------------
__global__ void k_scan2(float* __restrict__ chA, const float* __restrict__ chH) {
  int gid = blockIdx.x * 256 + threadIdx.x;   // BB*DI
  int d = gid & (DI - 1);
  int b = gid >> 9;
  float hrun[16];
#pragma unroll
  for (int n = 0; n < 16; ++n) hrun[n] = 0.f;
  for (int c = 0; c < NCH; ++c) {
    size_t base = (((size_t)b * NCH + c) * DI + d) * 16;
#pragma unroll
    for (int n = 0; n < 16; ++n) {
      float a  = chA[base + n];
      float hl = chH[base + n];
      chA[base + n] = hrun[n];        // h at chunk start
      hrun[n] = a * hrun[n] + hl;
    }
  }
}

// ---------------- scan pass 3: replay with h0, emit y = h.C + D*x ----------------
__global__ void k_scan3(const float* __restrict__ dt, const float* __restrict__ xs,
                        const float* __restrict__ bc, const float* __restrict__ A_log,
                        const float* __restrict__ Dp, const float* __restrict__ chA,
                        float* __restrict__ yout) {
  int gid = blockIdx.x * 256 + threadIdx.x;
  int d  = gid & (DI - 1);
  int t2 = gid >> 9;
  int ck = t2 & (NCH - 1);
  int b  = t2 >> 6;
  float Ac[16], hst[16];
  size_t base = (((size_t)b * NCH + ck) * DI + d) * 16;
#pragma unroll
  for (int n = 0; n < 16; ++n) {
    Ac[n]  = -__expf(A_log[d * DS + n]);
    hst[n] = chA[base + n];
  }
  float Dv = Dp[d];
  int t0 = ck * CH;
  for (int tt = 0; tt < CH; ++tt) {
    size_t row = (size_t)b * LL + t0 + tt;
    float dtv = dt[row * DI + d];
    float xv  = xs[row * DI + d];
    float dtx = dtv * xv;
    const float* Bp = bc + row * 32;
    float y = 0.f;
#pragma unroll
    for (int n = 0; n < 16; ++n) {
      float e = __expf(Ac[n] * dtv);
      hst[n] = e * hst[n] + dtx * Bp[n];
      y += hst[n] * Bp[16 + n];
    }
    yout[row * DI + d] = y + Dv * xv;
  }
}

// ---------------- gating: g = y * silu(z) -> bf16 ----------------
__global__ void k_gate(const float* __restrict__ y, const float* __restrict__ xz,
                       bf16* __restrict__ g) {
  size_t idx = (size_t)blockIdx.x * 256 + threadIdx.x;   // MR*DI
  int d = idx & (DI - 1);
  size_t row = idx >> 9;
  float z = xz[row * (2 * DI) + DI + d];
  float sil = z / (1.f + __expf(-z));
  g[idx] = (bf16)(y[idx] * sil);
}

// ---------------- final LN + mean over L ----------------
__global__ void k_final(const float* __restrict__ h, const float* __restrict__ g,
                        const float* __restrict__ bta, float* __restrict__ out) {
  __shared__ float sacc[DM];
  int tid = threadIdx.x;
  sacc[tid] = 0.f;
  __syncthreads();
  int wave = tid >> 5, lane = tid & 31;
  int b = blockIdx.y;
  int rowBase = blockIdx.x * 128 + wave * 16;
  float acc[8];
#pragma unroll
  for (int j = 0; j < 8; ++j) acc[j] = 0.f;
  for (int rr = 0; rr < 16; ++rr) {
    size_t row = (size_t)b * LL + rowBase + rr;
    const float* p = h + row * DM;
    float v[8], s = 0.f, ss = 0.f;
#pragma unroll
    for (int j = 0; j < 8; ++j) {
      v[j] = p[j * 32 + lane];
      s += v[j]; ss += v[j] * v[j];
    }
#pragma unroll
    for (int off = 16; off > 0; off >>= 1) {
      s  += __shfl_xor(s,  off, 32);
      ss += __shfl_xor(ss, off, 32);
    }
    float mean = s * (1.f / DM);
    float var  = ss * (1.f / DM) - mean * mean;
    float rinv = rsqrtf(var + 1e-5f);
#pragma unroll
    for (int j = 0; j < 8; ++j) {
      int c = j * 32 + lane;
      acc[j] += (v[j] - mean) * rinv * g[c] + bta[c];
    }
  }
#pragma unroll
  for (int j = 0; j < 8; ++j) atomicAdd(&sacc[j * 32 + lane], acc[j]);
  __syncthreads();
  atomicAdd(&out[b * DM + tid], sacc[tid] * (1.f / LL));
}

// ---------------- host side ----------------
extern "C" void kernel_launch(void* const* d_in, const int* in_sizes, int n_in,
                              void* d_out, int out_size, void* d_ws, size_t ws_size,
                              hipStream_t stream) {
  (void)in_sizes; (void)n_in; (void)out_size; (void)ws_size;
  const float* x         = (const float*)d_in[0];
  const float* in_w      = (const float*)d_in[1];
  const float* in_b      = (const float*)d_in[2];
  const float* in_norm_g = (const float*)d_in[3];
  const float* in_norm_b = (const float*)d_in[4];
  const float* norm_g    = (const float*)d_in[5];
  const float* norm_b    = (const float*)d_in[6];
  const float* inproj_w  = (const float*)d_in[7];
  const float* conv_w    = (const float*)d_in[8];
  const float* conv_b    = (const float*)d_in[9];
  const float* dt_w      = (const float*)d_in[10];
  const float* dt_b      = (const float*)d_in[11];
  const float* Bp_w      = (const float*)d_in[12];
  const float* Cp_w      = (const float*)d_in[13];
  const float* A_log     = (const float*)d_in[14];
  const float* D_param   = (const float*)d_in[15];
  const float* outproj_w = (const float*)d_in[16];
  const float* normf_g   = (const float*)d_in[17];
  const float* normf_b   = (const float*)d_in[18];
  float* out = (float*)d_out;

  // workspace layout
  char* ws = (char*)d_ws;
  size_t off = 0;
  auto alloc = [&](size_t bytes) -> char* {
    char* p = ws + off;
    off = (off + bytes + 255) & ~(size_t)255;
    return p;
  };
  float* h    = (float*)alloc((size_t)MR * DM * 4);
  bf16*  hn   = (bf16*) alloc((size_t)MR * DM * 2);
  float* xz   = (float*)alloc((size_t)MR * 2 * DI * 4);
  float* xs   = (float*)alloc((size_t)MR * DI * 4);
  bf16*  xsb  = (bf16*) alloc((size_t)MR * DI * 2);   // reused as gated bf16
  float* dtb  = (float*)alloc((size_t)MR * DI * 4);
  float* bcb  = (float*)alloc((size_t)MR * 32 * 4);
  float* yb   = (float*)alloc((size_t)MR * DI * 4);
  float* chA  = (float*)alloc((size_t)BB * NCH * DI * 16 * 4);
  float* chH  = (float*)alloc((size_t)BB * NCH * DI * 16 * 4);
  const size_t FR_IN  = (size_t)(DM / 32) * (2 * DI / 16) * 512;   // 262144
  const size_t FR_DT  = (size_t)(DI / 32) * (DI / 16) * 512;       // 262144
  const size_t FR_BC1 = (size_t)(DI / 32) * (DS / 16) * 512;       // 8192 (one of Bp/Cp)
  const size_t FR_OUT = (size_t)(DI / 32) * (DM / 16) * 512;       // 131072
  bf16* wf_in  = (bf16*)alloc(NLAY * FR_IN * 2);
  bf16* wf_dt  = (bf16*)alloc(NLAY * FR_DT * 2);
  bf16* wf_bc  = (bf16*)alloc(NLAY * 2 * FR_BC1 * 2);
  bf16* wf_out = (bf16*)alloc(NLAY * FR_OUT * 2);

  dim3 blk(256);
  auto packGrid = [](size_t frags) { return dim3((unsigned)((frags * 32 + 255) / 256)); };

  // pack all weights (bf16 fragment layout), all layers
  for (int i = 0; i < NLAY; ++i) {
    hipLaunchKernelGGL(k_pack_w, packGrid((DM/32)*(2*DI/16)), blk, 0, stream,
                       inproj_w + (size_t)i * DM * 2 * DI, wf_in + i * FR_IN, DM, 2 * DI);
    hipLaunchKernelGGL(k_pack_w, packGrid((DI/32)*(DI/16)), blk, 0, stream,
                       dt_w + (size_t)i * DI * DI, wf_dt + i * FR_DT, DI, DI);
    hipLaunchKernelGGL(k_pack_w, packGrid((DI/32)*(DS/16)), blk, 0, stream,
                       Bp_w + (size_t)i * DI * DS, wf_bc + i * 2 * FR_BC1, DI, DS);
    hipLaunchKernelGGL(k_pack_w, packGrid((DI/32)*(DS/16)), blk, 0, stream,
                       Cp_w + (size_t)i * DI * DS, wf_bc + i * 2 * FR_BC1 + FR_BC1, DI, DS);
    hipLaunchKernelGGL(k_pack_w, packGrid((DI/32)*(DM/16)), blk, 0, stream,
                       outproj_w + (size_t)i * DI * DM, wf_out + i * FR_OUT, DI, DM);
  }

  // embed + input LN (in-place f32)
  hipLaunchKernelGGL(k_embed, dim3((unsigned)((size_t)MR * DM / 256)), blk, 0, stream,
                     x, in_w, in_b, h);
  hipLaunchKernelGGL(k_layernorm, dim3(MR / 8), blk, 0, stream,
                     h, in_norm_g, in_norm_b, h, (bf16*)nullptr);

  for (int i = 0; i < NLAY; ++i) {
    // LN -> bf16 activations
    hipLaunchKernelGGL(k_layernorm, dim3(MR / 8), blk, 0, stream,
                       h, norm_g + i * DM, norm_b + i * DM, (float*)nullptr, hn);
    // inproj: [M,256] x [256,1024] -> xz
    hipLaunchKernelGGL((k_gemm_wmma<4, 0>), dim3((2 * DI) / 64, MR / 128), blk, 0, stream,
                       hn, wf_in + i * FR_IN, xz, (const float*)nullptr, 2 * DI, DM);
    // causal dwconv + SiLU
    hipLaunchKernelGGL(k_conv_silu, dim3((unsigned)((size_t)MR * DI / 256)), blk, 0, stream,
                       xz, conv_w + (size_t)i * DI * DC, conv_b + i * DI, xs, xsb);
    // dt: [M,512] x [512,512], +bias, softplus
    hipLaunchKernelGGL((k_gemm_wmma<4, 1>), dim3(DI / 64, MR / 128), blk, 0, stream,
                       xsb, wf_dt + i * FR_DT, dtb, dt_b + i * DI, DI, DI);
    // B|C: [M,512] x [512,32]
    hipLaunchKernelGGL((k_gemm_wmma<2, 0>), dim3(1, MR / 128), blk, 0, stream,
                       xsb, wf_bc + i * 2 * FR_BC1, bcb, (const float*)nullptr, 32, DI);
    // chunked selective scan
    hipLaunchKernelGGL(k_scan1, dim3(BB * NCH * DI / 256), blk, 0, stream,
                       dtb, xs, bcb, A_log + (size_t)i * DI * DS, chA, chH);
    hipLaunchKernelGGL(k_scan2, dim3(BB * DI / 256), blk, 0, stream, chA, chH);
    hipLaunchKernelGGL(k_scan3, dim3(BB * NCH * DI / 256), blk, 0, stream,
                       dtb, xs, bcb, A_log + (size_t)i * DI * DS,
                       D_param + i * DI, chA, yb);
    // gate: g = y * silu(z)  (reuse xsb)
    hipLaunchKernelGGL(k_gate, dim3((unsigned)((size_t)MR * DI / 256)), blk, 0, stream,
                       yb, xz, xsb);
    // outproj: [M,512] x [512,256], h += result
    hipLaunchKernelGGL((k_gemm_wmma<4, 2>), dim3(DM / 64, MR / 128), blk, 0, stream,
                       xsb, wf_out + i * FR_OUT, h, (const float*)nullptr, DM, DI);
  }

  // final LN + mean over L
  hipMemsetAsync(d_out, 0, (size_t)BB * DM * sizeof(float), stream);
  hipLaunchKernelGGL(k_final, dim3(LL / 128, BB), blk, 0, stream,
                     h, normf_g, normf_b, out);
}
